// GAT_9259949490983
// MI455X (gfx1250) — compile-verified, matrix operand order checked by reading, not run on hardware
//
#include <hip/hip_runtime.h>

typedef __attribute__((ext_vector_type(8)))  float  v8f;
typedef __attribute__((ext_vector_type(8)))  __bf16 bf16x8;
typedef __attribute__((ext_vector_type(16))) __bf16 v16bf;

#define BATCH   32
#define C_IN    192
#define T_DIM   256
#define V_N     21
#define HEADS   3
#define C_OUTD  192
#define HC      576                 // HEADS*C_OUTD
#define S_TV    (T_DIM * V_N)       // 5376
#define NTOT    (BATCH * S_TV)      // 172032 nodes
#define NODES_PER_WG 63             // 3 (b,t) slices * 21 vertices
#define XS_BSTRIDE 200              // bf16 elems per m-row (192 + 8 pad; 400B, 16B-mult)
#define PS_STRIDE  580              // 576 + 4 pad (8*580 mod 64 = 32 -> halves disjoint)
#define NWAVES  16
#define BLOCK   512
#define W_ELEMS (HC * C_IN)         // 110592

// ---- RNE float <-> bf16 helpers (explicit, no reliance on cast semantics) ----
static __device__ __forceinline__ __bf16 f2bf(float f) {
    union { float f; unsigned u; } v; v.f = f;
    unsigned r = v.u + 0x7FFFu + ((v.u >> 16) & 1u);
    union { unsigned short s; __bf16 b; } o; o.s = (unsigned short)(r >> 16);
    return o.b;
}
static __device__ __forceinline__ float bf2f(__bf16 b) {
    union { unsigned short s; __bf16 b; } i; i.b = b;
    union { unsigned u; float f; } o; o.u = ((unsigned)i.s) << 16;
    return o.f;
}

// Load one 16x16x32 fragment for this lane: K = kb+{0..7} and kb+16+{0..7},
// both chunks 16B-aligned and contiguous -> two b128 loads, no shuffling.
template <typename PT>
static __device__ __forceinline__ v16bf load_frag(const PT* p) {
    bf16x8 a = *reinterpret_cast<const bf16x8*>(p);
    bf16x8 b = *reinterpret_cast<const bf16x8*>(p + 16);
    return __builtin_shufflevector(a, b, 0,1,2,3,4,5,6,7,8,9,10,11,12,13,14,15);
}

// ---- pre-kernel: split W (fp32) into bf16 hi/lo halves in workspace ----
__global__ void wsplit_kernel(const float* __restrict__ W,
                              __bf16* __restrict__ Whi,
                              __bf16* __restrict__ Wlo) {
    int i = blockIdx.x * blockDim.x + threadIdx.x;
    if (i < W_ELEMS) {
        float w = W[i];
        __bf16 h = f2bf(w);
        Whi[i] = h;
        Wlo[i] = f2bf(w - bf2f(h));
    }
}

__global__ __launch_bounds__(BLOCK, 1)
void gat_fused_kernel(const float*  __restrict__ x,      // (B, C_IN, T, V)
                      const float*  __restrict__ Adj,    // (H, V, V)
                      const __bf16* __restrict__ Whi,    // (HC, C_IN) bf16 hi
                      const __bf16* __restrict__ Wlo,    // (HC, C_IN) bf16 lo
                      const float*  __restrict__ ssrc,   // (1, H, C)
                      const float*  __restrict__ stgt,   // (1, H, C)
                      float* __restrict__ out)           // (NTOT, C) flat
{
    __shared__ __align__(16) __bf16 XsH[64 * XS_BSTRIDE];   // [m][k] hi  25.6 KB
    __shared__ __align__(16) __bf16 XsL[64 * XS_BSTRIDE];   // [m][k] lo  25.6 KB
    __shared__ float Ps[64 * PS_STRIDE];                    // [m][hc]   148.5 KB
    __shared__ float ssL[64 * HEADS];
    __shared__ float stL[64 * HEADS];
    __shared__ float attnL[9 * 21 * 22];
    __shared__ float rsum[63 * 24];

    const int tid  = threadIdx.x;
    const int wave = tid >> 5;
    const int lane = tid & 31;
    const int ml   = lane & 15;
    const int half = lane >> 4;

    const int baseNode = blockIdx.x * NODES_PER_WG;
    const int nNodes   = min(NODES_PER_WG, NTOT - baseNode);

    // ---------------- Phase 1: gather/transpose X tile, split into bf16 hi/lo --
    for (int idx = tid; idx < C_IN * 64; idx += BLOCK) {
        int k = idx >> 6;
        int m = idx & 63;
        float val = 0.0f;
        if (m < nNodes) {
            int n   = baseNode + m;
            int b   = n / S_TV;
            int rem = n - b * S_TV;
            val = x[(size_t)b * (C_IN * S_TV) + (size_t)k * S_TV + rem];
        }
        __bf16 h = f2bf(val);
        XsH[m * XS_BSTRIDE + k] = h;
        XsL[m * XS_BSTRIDE + k] = f2bf(val - bf2f(h));
    }
    __syncthreads();

    // ---------------- Phase 2: 3xBF16 split GEMM via V_WMMA_F32_16X16X32_BF16 --
    // Ps[m][nc] = sum_k X[m][k] * W[nc][k];  M=64 (4 tiles), N=576 (36 tiles), K=192
    // NOTE: unroll-1 on nt/kt is load-bearing: full unroll made LLVM hoist all
    // 48 A-fragments (~384 VGPRs) out of the nt loop and spill them to scratch.
    #pragma unroll 1
    for (int nt = wave; nt < 36; nt += NWAVES) {
        v8f acc[4] = {v8f{}, v8f{}, v8f{}, v8f{}};
        const __bf16* whrow = Whi + (size_t)(nt * 16 + ml) * C_IN;  // lane's B column
        const __bf16* wlrow = Wlo + (size_t)(nt * 16 + ml) * C_IN;
        #pragma unroll 1
        for (int kt = 0; kt < 6; ++kt) {
            const int kb = kt * 32 + 8 * half;      // lane-half K base
            v16bf bh = load_frag(whrow + kb);
            v16bf bl = load_frag(wlrow + kb);
            #pragma unroll
            for (int mt = 0; mt < 4; ++mt) {
                v16bf ah = load_frag(&XsH[(mt * 16 + ml) * XS_BSTRIDE + kb]);
                v16bf al = load_frag(&XsL[(mt * 16 + ml) * XS_BSTRIDE + kb]);
                acc[mt] = __builtin_amdgcn_wmma_f32_16x16x32_bf16(false, ah, false, bh, (short)0, acc[mt], false, false);
                acc[mt] = __builtin_amdgcn_wmma_f32_16x16x32_bf16(false, al, false, bh, (short)0, acc[mt], false, false);
                acc[mt] = __builtin_amdgcn_wmma_f32_16x16x32_bf16(false, ah, false, bl, (short)0, acc[mt], false, false);
            }
        }
        // D layout: VGPR r -> M = r + 8*half, N = ml
        int col = nt * 16 + ml;
        #pragma unroll
        for (int mt = 0; mt < 4; ++mt)
            #pragma unroll
            for (int r = 0; r < 8; ++r)
                Ps[(mt * 16 + r + 8 * half) * PS_STRIDE + col] = acc[mt][r];
    }
    __syncthreads();

    // ---------------- Phase 3: attention scalars ss/st per (node, head) --------
    for (int p = wave; p < NODES_PER_WG * HEADS; p += NWAVES) {
        int m = p / HEADS;
        int h = p - m * HEADS;
        const float* pr = &Ps[m * PS_STRIDE + h * C_OUTD];
        const float* w1 = ssrc + h * C_OUTD;
        const float* w2 = stgt + h * C_OUTD;
        float s1 = 0.0f, s2 = 0.0f;
        for (int c = lane; c < C_OUTD; c += 32) {
            float v = pr[c];
            s1 += v * w1[c];
            s2 += v * w2[c];
        }
        #pragma unroll
        for (int off = 16; off > 0; off >>= 1) {
            s1 += __shfl_xor(s1, off, 32);
            s2 += __shfl_xor(s2, off, 32);
        }
        if (lane == 0) { ssL[p] = s1; stL[p] = s2; }
    }
    __syncthreads();

    // ---------------- Phase 4: masked softmax per unit (local_bp, head) --------
    if (wave < 9) {
        const int u   = wave;
        const int lbp = u / 3;
        const int h   = u - lbp * 3;
        if (lane < V_N) {
            int i = lane;
            float si = ssL[(lbp * V_N + i) * HEADS + h];
            float sv[V_N];
            float mx = -3.4e38f;
            #pragma unroll
            for (int j = 0; j < V_N; ++j) {
                float a = Adj[(h * V_N + i) * V_N + j];
                float s;
                if (a != 0.0f) {
                    float z = si + stL[(lbp * V_N + j) * HEADS + h];
                    z = (z > 0.0f) ? z : 0.2f * z;      // leaky relu
                    s = z + a;                          // + A_masked (==a where kept)
                } else {
                    s = -3.4e38f;                       // -inf mask
                }
                sv[j] = s;
                mx = fmaxf(mx, s);
            }
            float den = 0.0f;
            #pragma unroll
            for (int j = 0; j < V_N; ++j) {
                float e = (sv[j] > -1.0e37f) ? expf(sv[j] - mx) : 0.0f;
                sv[j] = e;
                den += e;
            }
            float inv = 1.0f / den;
            #pragma unroll
            for (int j = 0; j < V_N; ++j)
                attnL[(u * V_N + i) * 22 + j] = sv[j] * inv;
        }
    }
    __syncthreads();

    // ---------------- Phase 5: pre-sum attn rows over output triples -----------
    for (int idx = tid; idx < 9 * 7 * V_N; idx += BLOCK) {
        int j  = idx % V_N;
        int gq = idx / V_N;                 // == g_local in [0,63)
        int u  = gq / 7;
        int q  = gq - u * 7;
        float r = attnL[(u * V_N + 3 * q + 0) * 22 + j]
                + attnL[(u * V_N + 3 * q + 1) * 22 + j]
                + attnL[(u * V_N + 3 * q + 2) * 22 + j];
        rsum[gq * 24 + j] = r;
    }
    __syncthreads();

    // ---------------- Phase 6: aggregate + skip + mean, write output -----------
    for (int idx = tid; idx < NODES_PER_WG * C_OUTD; idx += BLOCK) {
        int c = idx % C_OUTD;
        int g = idx / C_OUTD;
        if (baseNode + g >= NTOT) continue;
        int u   = g / 7;
        int lbp = u / 3;
        int h   = u - lbp * 3;
        const float* rs = &rsum[g * 24];
        float acc = 0.0f;
        #pragma unroll
        for (int j = 0; j < V_N; ++j)
            acc += rs[j] * Ps[(lbp * V_N + j) * PS_STRIDE + h * C_OUTD + c];
        float skip = Ps[g * PS_STRIDE + 0 * C_OUTD + c]
                   + Ps[g * PS_STRIDE + 1 * C_OUTD + c]
                   + Ps[g * PS_STRIDE + 2 * C_OUTD + c];
        out[(size_t)(baseNode + g) * C_OUTD + c] = (acc + skip) * (1.0f / 3.0f);
    }
}

extern "C" void kernel_launch(void* const* d_in, const int* in_sizes, int n_in,
                              void* d_out, int out_size, void* d_ws, size_t ws_size,
                              hipStream_t stream) {
    const float* x    = (const float*)d_in[0];   // (32,192,256,21)
    const float* Adj  = (const float*)d_in[1];   // (3,21,21)
    const float* W    = (const float*)d_in[2];   // (576,192)
    const float* ssrc = (const float*)d_in[3];   // (1,3,192)
    const float* stgt = (const float*)d_in[4];   // (1,3,192)
    float* out        = (float*)d_out;           // (172032,192) flat

    (void)in_sizes; (void)n_in; (void)out_size; (void)ws_size;

    __bf16* Whi = (__bf16*)d_ws;                 // 110592 bf16
    __bf16* Wlo = Whi + W_ELEMS;                 // 110592 bf16 (ws usage: 442 KB)

    wsplit_kernel<<<(W_ELEMS + 255) / 256, 256, 0, stream>>>(W, Whi, Wlo);

    const int nWG = (NTOT + NODES_PER_WG - 1) / NODES_PER_WG;  // 2731
    gat_fused_kernel<<<nWG, BLOCK, 0, stream>>>(x, Adj, Whi, Wlo, ssrc, stgt, out);
}
